// MoE_60481729462675
// MI455X (gfx1250) — matrix operand order repaired and checked by reading, not compiled, not run on hardware
//
#include <hip/hip_runtime.h>
#include <math.h>
#include <stdint.h>

// ---------------------------------------------------------------------------
// MoE block for MI455X (gfx1250): bf16 WMMA GEMMs, hard-gated grouped experts,
// double-buffered LDS pipeline, TDM (tensor_load_to_lds) for bf16 A-tiles.
// ---------------------------------------------------------------------------

typedef __attribute__((ext_vector_type(16))) __bf16 v16bf;
typedef __attribute__((ext_vector_type(8)))  float  v8f;
typedef __attribute__((ext_vector_type(4)))  unsigned int u32x4;
typedef __attribute__((ext_vector_type(8)))  int i32x8;
typedef __attribute__((ext_vector_type(4)))  int i32x4;

#define BM 128
#define BN 64
#define BK 32
#define ASTR 40   // LDS row stride (bf16): 80B rows -> 16B aligned ds_load_b128
#define BSTR 40

#if defined(__gfx1250__) && __has_builtin(__builtin_amdgcn_tensor_load_to_lds)
#define HAVE_TDM 1
#else
#define HAVE_TDM 0
#endif

#if HAVE_TDM
#if __clang_major__ >= 23
#define TDM_CALL(g0, g1, z4a, z4b) \
  do { i32x8 _z8 = {0,0,0,0,0,0,0,0}; \
       __builtin_amdgcn_tensor_load_to_lds((g0), (g1), (z4a), (z4b), _z8, 0); } while (0)
#else
#define TDM_CALL(g0, g1, z4a, z4b) \
  __builtin_amdgcn_tensor_load_to_lds((g0), (g1), (z4a), (z4b), 0)
#endif
#endif

// pack two fp32 -> packed bf16x2 with a single v_perm_b32 (truncate)
__device__ __forceinline__ unsigned pkbf(float a, float b) {
  return __builtin_amdgcn_perm(__builtin_bit_cast(unsigned, b),
                               __builtin_bit_cast(unsigned, a), 0x07060302u);
}
__device__ __forceinline__ unsigned short bfh(float f) {
  return (unsigned short)(__builtin_bit_cast(unsigned, f) >> 16);
}

union Frag { v16bf v; uint4 q[2]; };

__device__ __forceinline__ float warp_sum(float p) {
  for (int o = 16; o; o >>= 1) p += __shfl_xor(p, o, 32);
  return p;
}
__device__ __forceinline__ float warp_max(float p) {
  for (int o = 16; o; o >>= 1) p = fmaxf(p, __shfl_xor(p, o, 32));
  return p;
}
__device__ __forceinline__ float gumbel_hash(unsigned s) {
  s ^= s >> 16; s *= 0x7feb352du; s ^= s >> 15; s *= 0x846ca68bu; s ^= s >> 16;
  float u = (float)s * 2.3283064e-10f;
  u = u * 0.99999988f + 1e-7f;
  return -logf(-logf(u));
}

// ---------------------------------------------------------------------------
// OUT[M,N] = epi( A[M,K] * W[N,K]^T )    (bf16 WMMA, f32 accumulate)
//   DUAL : two B matrices (W1,W3), two accumulator sets
//   EPI  : 0 = f32 out (+bias), 1 = bf16 out silu(acc0)*acc1, 2 = scatter+add
//   APERM: gather A rows through perm[] (padded grouped GEMM, -1 => zero row)
//   ABF16: A is bf16 (no conversion; staged by TDM when available)
// A/B: element s of lane half h is k = 16h + s (wave32); C/D per CDNA5 ISA 7.12.2: reg r -> row r+8*half.
// ---------------------------------------------------------------------------
template <bool DUAL, int EPI, bool APERM, bool ABF16>
__global__ __launch_bounds__(256) void gemm_kernel(
    const void* __restrict__ Avoid, const float* __restrict__ W0,
    const float* __restrict__ W1, const float* __restrict__ bias,
    const float* __restrict__ addsrc, void* __restrict__ OUTv,
    int Mvalid, int N, int K, int ldo,
    const int* __restrict__ perm, const int* __restrict__ texp,
    long long expStride) {
  __shared__ __align__(16) unsigned short sA[2][BM * ASTR];
  __shared__ __align__(16) unsigned short sB0[2][BN * BSTR];
  __shared__ __align__(16) unsigned short sB1[2][BN * BSTR];

  const float* Af = (const float*)Avoid;
  const unsigned short* Ah = (const unsigned short*)Avoid;

  const int tid = threadIdx.x;
  const int ntile = blockIdx.x, mtile = blockIdx.y;
  const int e = texp ? texp[mtile] : 0;
  const float* Wa = W0 + (long long)e * expStride;
  const float* Wb = DUAL ? (W1 + (long long)e * expStride) : W0;

  // ---- staging coordinates
  const int ar = tid >> 1;                 // A fp32: 2 thr/row x 16 cols
  const int acs = (tid & 1) << 4;
  const int agrow = mtile * BM + ar;
  int arow;
  if (APERM) arow = perm[agrow];
  else       arow = (agrow < Mvalid) ? agrow : -1;
  const float* aptr = (arow >= 0) ? (Af + (size_t)arow * K + acs) : nullptr;

  const int bn = tid >> 2;                 // B: 4 thr/row x 8 cols
  const int bcs = (tid & 3) << 3;
  const float* b0ptr = Wa + (size_t)(ntile * BN + bn) * K + bcs;
  const float* b1ptr = Wb + (size_t)(ntile * BN + bn) * K + bcs;

  const int lane = tid & 31, wave = tid >> 5;
  const int half = lane >> 4, l16 = lane & 15;
  const int wm = wave >> 1, wn = wave & 1;  // 4x2 wave grid -> 32x32 per wave

  v8f zero;
#pragma unroll
  for (int r = 0; r < 8; r++) zero[r] = 0.0f;
  v8f acc0[2][2], acc1[2][2];
#pragma unroll
  for (int i = 0; i < 2; i++)
#pragma unroll
    for (int j = 0; j < 2; j++) { acc0[i][j] = zero; acc1[i][j] = zero; }

  // ---- register prefetch state
  float4 afr[4];          // A fp32 tile slice
  uint4  ahr[2];          // A bf16 tile slice (non-TDM fallback)
  float4 b0r[2], b1r[2];  // B tile slices

  auto load_a32 = [&](int k0) {
    if (aptr) {
      const float4* p = (const float4*)(aptr + k0);
      afr[0] = p[0]; afr[1] = p[1]; afr[2] = p[2]; afr[3] = p[3];
      if (k0 + BK < K) __builtin_prefetch(aptr + k0 + BK, 0, 0);
    } else {
      float4 z = {0, 0, 0, 0};
      afr[0] = z; afr[1] = z; afr[2] = z; afr[3] = z;
    }
  };
  auto store_a32 = [&](int cur) {
    uint4 q0 = {pkbf(afr[0].x, afr[0].y), pkbf(afr[0].z, afr[0].w),
                pkbf(afr[1].x, afr[1].y), pkbf(afr[1].z, afr[1].w)};
    uint4 q1 = {pkbf(afr[2].x, afr[2].y), pkbf(afr[2].z, afr[2].w),
                pkbf(afr[3].x, afr[3].y), pkbf(afr[3].z, afr[3].w)};
    *(uint4*)&sA[cur][ar * ASTR + acs] = q0;
    *(uint4*)&sA[cur][ar * ASTR + acs + 8] = q1;
  };
  auto load_a16 = [&](int k0) {   // fallback path when no TDM builtin
#pragma unroll
    for (int j = 0; j < 2; j++) {
      const int c = tid * 2 + j;
      const int r = c >> 2, col = (c & 3) * 8;
      const int grow = mtile * BM + r;
      if (grow < Mvalid) {
        ahr[j] = *(const uint4*)(Ah + (size_t)grow * K + k0 + col);
      } else {
        uint4 z = {0, 0, 0, 0}; ahr[j] = z;
      }
    }
  };
  auto store_a16 = [&](int cur) {
#pragma unroll
    for (int j = 0; j < 2; j++) {
      const int c = tid * 2 + j;
      const int r = c >> 2, col = (c & 3) * 8;
      *(uint4*)&sA[cur][r * ASTR + col] = ahr[j];
    }
  };
  auto load_b = [&](int k0) {
    const float4* p = (const float4*)(b0ptr + k0);
    b0r[0] = p[0]; b0r[1] = p[1];
    if (k0 + BK < K) __builtin_prefetch(b0ptr + k0 + BK, 0, 0);
    if (DUAL) {
      const float4* r = (const float4*)(b1ptr + k0);
      b1r[0] = r[0]; b1r[1] = r[1];
    }
  };
  auto store_b = [&](int cur) {
    uint4 q = {pkbf(b0r[0].x, b0r[0].y), pkbf(b0r[0].z, b0r[0].w),
               pkbf(b0r[1].x, b0r[1].y), pkbf(b0r[1].z, b0r[1].w)};
    *(uint4*)&sB0[cur][bn * BSTR + bcs] = q;
    if (DUAL) {
      uint4 q2 = {pkbf(b1r[0].x, b1r[0].y), pkbf(b1r[0].z, b1r[0].w),
                  pkbf(b1r[1].x, b1r[1].y), pkbf(b1r[1].z, b1r[1].w)};
      *(uint4*)&sB1[cur][bn * BSTR + bcs] = q2;
    }
  };

#if HAVE_TDM
  // TDM: 2D tile [128 x 32] bf16 from A into LDS, with 4-DWORD pad per row
  // (16 DWORDs) to reproduce the 80-byte padded LDS stride (ASTR).
  auto tdm_issue = [&](int k0, int buf) {
    unsigned long long ga = (unsigned long long)(uintptr_t)Ah +
        ((unsigned long long)mtile * BM * (unsigned)K + (unsigned)k0) * 2ull;
    unsigned ldsoff = (unsigned)(uintptr_t)(&sA[buf][0]);
    u32x4 g0;
    g0[0] = 1u;                                   // count=1, user mode
    g0[1] = ldsoff;                               // LDS byte address
    g0[2] = (unsigned)ga;                         // global_addr[31:0]
    g0[3] = (unsigned)(ga >> 32) | (2u << 30);    // addr[56:32] | type=2
    i32x8 g1;
    g1[0] = (1 << 16) | (1 << 20) | (3 << 22) | (3 << 25); // 2B, pad 4DW/16DW
    g1[1] = (int)(((unsigned)K & 0xFFFFu) << 16);          // tensor_dim0 lo
    g1[2] = (int)(((unsigned)K >> 16) |
                  (((unsigned)Mvalid & 0xFFFFu) << 16));   // dim0 hi | dim1 lo
    g1[3] = (int)(((unsigned)Mvalid >> 16) | (32u << 16)); // dim1 hi | tile_dim0
    g1[4] = 128;                                           // tile_dim1, dim2=0
    g1[5] = K;                                             // dim0_stride lo32
    g1[6] = 0;
    g1[7] = 0;
    i32x4 z4 = {0, 0, 0, 0};
    TDM_CALL(g0, g1, z4, z4);
  };
#endif

  auto compute = [&](int cur) {
    Frag a[2], b0[2], b1[2];
#pragma unroll
    for (int mi = 0; mi < 2; mi++) {
      const int row = wm * 32 + mi * 16 + l16;
      a[mi].q[0] = *(const uint4*)&sA[cur][row * ASTR + half * 16];
      a[mi].q[1] = *(const uint4*)&sA[cur][row * ASTR + half * 16 + 8];
    }
#pragma unroll
    for (int ni = 0; ni < 2; ni++) {
      const int col = wn * 32 + ni * 16 + l16;
      b0[ni].q[0] = *(const uint4*)&sB0[cur][col * BSTR + half * 16];
      b0[ni].q[1] = *(const uint4*)&sB0[cur][col * BSTR + half * 16 + 8];
      if (DUAL) {
        b1[ni].q[0] = *(const uint4*)&sB1[cur][col * BSTR + half * 16];
        b1[ni].q[1] = *(const uint4*)&sB1[cur][col * BSTR + half * 16 + 8];
      }
    }
#pragma unroll
    for (int mi = 0; mi < 2; mi++)
#pragma unroll
      for (int ni = 0; ni < 2; ni++) {
        acc0[mi][ni] = __builtin_amdgcn_wmma_f32_16x16x32_bf16(
            false, a[mi].v, false, b0[ni].v, (short)0, acc0[mi][ni], false, false);
        if (DUAL)
          acc1[mi][ni] = __builtin_amdgcn_wmma_f32_16x16x32_bf16(
              false, a[mi].v, false, b1[ni].v, (short)0, acc1[mi][ni], false, false);
      }
  };

  // ---- software-pipelined, double-buffered main loop (1 barrier / K-step)
  const int niter = K / BK;
  if (!ABF16) load_a32(0);
#if HAVE_TDM
  if (ABF16 && wave == 0) tdm_issue(0, 0);
#else
  if (ABF16) load_a16(0);
#endif
  load_b(0);

  for (int i = 0; i < niter; i++) {
    const int cur = i & 1;
    const int kn = (i + 1) * BK;
    if (!ABF16) store_a32(cur);
#if !HAVE_TDM
    if (ABF16) store_a16(cur);
#endif
    store_b(cur);
    if (kn < K) {
      if (!ABF16) load_a32(kn);
#if !HAVE_TDM
      if (ABF16) load_a16(kn);
#endif
      load_b(kn);
    }
#if HAVE_TDM
    if (ABF16 && wave == 0) __builtin_amdgcn_s_wait_tensorcnt(0);
#endif
    __syncthreads();
#if HAVE_TDM
    if (ABF16 && wave == 0 && kn < K) tdm_issue(kn, cur ^ 1);
#endif
    compute(cur);
  }

  // ---- epilogue
#pragma unroll
  for (int mi = 0; mi < 2; mi++)
#pragma unroll
    for (int ni = 0; ni < 2; ni++) {
      const int gcol = ntile * BN + wn * 32 + ni * 16 + l16;
#pragma unroll
      for (int r = 0; r < 8; r++) {
        const int grow = mtile * BM + wm * 32 + mi * 16 + half * 8 + r;
        float v = acc0[mi][ni][r];
        if (EPI == 1) {          // h = silu(xW1^T)*(xW3^T), stored bf16
          const float g = v;
          v = (g / (1.0f + __expf(-g))) * acc1[mi][ni][r];
          if (grow < Mvalid)
            ((unsigned short*)OUTv)[(size_t)grow * ldo + gcol] = bfh(v);
        } else if (EPI == 2) {   // grouped-GEMM scatter + residual add
          const int orow = perm[grow];
          if (orow >= 0)
            ((float*)OUTv)[(size_t)orow * ldo + gcol] =
                addsrc[(size_t)orow * ldo + gcol] + v;
        } else {
          if (grow < Mvalid) {
            if (bias) v += bias[gcol];
            ((float*)OUTv)[(size_t)grow * ldo + gcol] = v;
          }
        }
      }
    }
}

// ---------------------------------------------------------------------------
// Attention over 77 caption keys: one wave32 per (token, head), hd=128
// ---------------------------------------------------------------------------
__global__ __launch_bounds__(256) void attn_kernel(
    const float* __restrict__ q, const float* __restrict__ kv,
    float* __restrict__ o) {
  const int tok = blockIdx.x;
  const int b = tok >> 11;
  const int h = threadIdx.x >> 5;
  const int lane = threadIdx.x & 31;
  const float4 qv = *(const float4*)(q + (size_t)tok * 1024 + h * 128 + lane * 4);
  const float scale = 0.08838834764831845f;  // 1/sqrt(128)
  float sc[3] = {-1e30f, -1e30f, -1e30f};
  const float* kbase = kv + (size_t)b * 77 * 2048 + h * 128 + lane * 4;
  for (int s = 0; s < 77; s++) {
    float4 k4 = *(const float4*)(kbase + (size_t)s * 2048);
    float p = qv.x * k4.x + qv.y * k4.y + qv.z * k4.z + qv.w * k4.w;
    p = warp_sum(p) * scale;
    if ((s & 31) == lane) sc[s >> 5] = p;
  }
  float m = warp_max(fmaxf(sc[0], fmaxf(sc[1], sc[2])));
  float es[3];
  es[0] = expf(sc[0] - m); es[1] = expf(sc[1] - m); es[2] = expf(sc[2] - m);
  const float inv = 1.0f / warp_sum(es[0] + es[1] + es[2]);
  float4 acc = {0, 0, 0, 0};
  const float* vbase = kv + (size_t)b * 77 * 2048 + 1024 + h * 128 + lane * 4;
  for (int s = 0; s < 77; s++) {
    float a = __shfl(es[s >> 5], s & 31, 32) * inv;
    float4 v4 = *(const float4*)(vbase + (size_t)s * 2048);
    acc.x += a * v4.x; acc.y += a * v4.y; acc.z += a * v4.z; acc.w += a * v4.w;
  }
  *(float4*)(o + (size_t)tok * 1024 + h * 128 + lane * 4) = acc;
}

// ---------------------------------------------------------------------------
// Gating / routing / loss helper kernels
// ---------------------------------------------------------------------------
__global__ void init_kernel(int* perm, int* counts, int* cursor) {
  int i = blockIdx.x * 256 + threadIdx.x;
  if (i < 4608) perm[i] = -1;
  if (i < 8) counts[i] = 0;
  if (i < 4) cursor[i] = 0;
}

__global__ __launch_bounds__(256) void lang_gate_kernel(
    const int* __restrict__ language, const float* __restrict__ lang_embed,
    const float* __restrict__ gw, const float* __restrict__ gb,
    int* __restrict__ lang_e, int* __restrict__ tileL, int* __restrict__ counts) {
  __shared__ float logit[2][4];
  const int w = threadIdx.x >> 5, lane = threadIdx.x & 31;
  const int b = w >> 2, ex = w & 3;
  const float* emb = lang_embed + (size_t)language[b] * 1024;
  const float* wt = gw + (size_t)ex * 1024;
  float p = 0.0f;
  for (int i = lane; i < 1024; i += 32) p += emb[i] * wt[i];
  p = warp_sum(p) + gb[ex] + gumbel_hash(0xABCD1234u ^ (unsigned)(b * 4 + ex));
  if (lane == 0) logit[b][ex] = p;
  __syncthreads();
  if (threadIdx.x < 2) {
    const int bb = threadIdx.x;
    float best = -1e30f; int bi = 0;
    for (int j = 0; j < 4; j++)
      if (logit[bb][j] > best) { best = logit[bb][j]; bi = j; }
    lang_e[bb] = bi;
    atomicAdd(&counts[bi], 2048);   // one-hot routing over T tokens
  }
  __syncthreads();
  if (threadIdx.x < 32) tileL[threadIdx.x] = lang_e[threadIdx.x >> 4];
}

__global__ __launch_bounds__(256) void cap_gate_kernel(
    const float* __restrict__ cross, const float* __restrict__ gw,
    const float* __restrict__ gb, int* __restrict__ cap_e,
    int* __restrict__ counts) {
  const int tok = blockIdx.x * 8 + (threadIdx.x >> 5);
  const int lane = threadIdx.x & 31;
  const float* xr = cross + (size_t)tok * 1024;
  float best = -1e30f; int bi = 0;
  for (int ex = 0; ex < 4; ex++) {
    const float* wt = gw + (size_t)ex * 1024;
    float p = 0.0f;
    for (int i = lane; i < 1024; i += 32) p += xr[i] * wt[i];
    p = warp_sum(p) + gb[ex] + gumbel_hash(0xC0FFEEu ^ (unsigned)(tok * 4 + ex));
    if (p > best) { best = p; bi = ex; }
  }
  if (lane == 0) { cap_e[tok] = bi; atomicAdd(&counts[4 + bi], 1); }
}

__global__ void prep_kernel(const int* counts, int* segoff, int* tileC) {
  if (threadIdx.x != 0 || blockIdx.x != 0) return;
  int off = 0;
  for (int ex = 0; ex < 4; ex++) {
    segoff[ex] = off;
    int seg = ((counts[4 + ex] + 127) >> 7) << 7;  // pad segments to 128 rows
    for (int t = off >> 7; t < (off + seg) >> 7; t++) tileC[t] = ex;
    off += seg;
  }
  segoff[4] = off;
  for (int t = off >> 7; t < 36; t++) tileC[t] = 0;
}

__global__ void scatter_kernel(const int* cap_e, const int* segoff,
                               int* cursor, int* perm) {
  int t = blockIdx.x * 256 + threadIdx.x;
  if (t >= 4096) return;
  int ex = cap_e[t];
  int pos = segoff[ex] + atomicAdd(&cursor[ex], 1);
  perm[pos] = t;
}

__global__ void loss_kernel(const int* counts, float* out_loss) {
  if (threadIdx.x == 0) {
    float loss = 0.0f;
    for (int i = 0; i < 8; i++) {
      float u = (float)counts[i] / 4096.0f;
      loss += u * logf(u + 1e-10f);
    }
    *out_loss = loss;
  }
}

// ---------------------------------------------------------------------------
extern "C" void kernel_launch(void* const* d_in, const int* in_sizes, int n_in,
                              void* d_out, int out_size, void* d_ws, size_t ws_size,
                              hipStream_t stream) {
  (void)in_sizes; (void)n_in; (void)out_size; (void)ws_size;
  const float* x        = (const float*)d_in[0];
  const float* caption  = (const float*)d_in[2];
  const int*   language = (const int*)d_in[3];
  const float* lang_emb = (const float*)d_in[4];
  const float* lgate_w  = (const float*)d_in[5];
  const float* lgate_b  = (const float*)d_in[6];
  const float* lw1      = (const float*)d_in[7];
  const float* lw2      = (const float*)d_in[8];
  const float* lw3      = (const float*)d_in[9];
  const float* in_w     = (const float*)d_in[10];
  const float* in_b     = (const float*)d_in[11];
  const float* out_w    = (const float*)d_in[12];
  const float* out_b    = (const float*)d_in[13];
  const float* cgate_w  = (const float*)d_in[14];
  const float* cgate_b  = (const float*)d_in[15];
  const float* cw1      = (const float*)d_in[16];
  const float* cw2      = (const float*)d_in[17];
  const float* cw3      = (const float*)d_in[18];
  float* out = (float*)d_out;

  // workspace layout
  char* w = (char*)d_ws;
  size_t o = 0;
  unsigned short* h_buf = (unsigned short*)(w + o); o += (size_t)4608 * 2816 * 2;
  float* first = (float*)(w + o); o += (size_t)4096 * 1024 * 4;
  float* qbuf  = (float*)(w + o); o += (size_t)4096 * 1024 * 4;
  float* attob = (float*)(w + o); o += (size_t)4096 * 1024 * 4;
  float* kvb   = (float*)(w + o); o += (size_t)256 * 2048 * 4;
  int* perm   = (int*)(w + o); o += 4608 * 4;
  int* cap_e  = (int*)(w + o); o += 4096 * 4;
  int* tileL  = (int*)(w + o); o += 32 * 4;
  int* tileC  = (int*)(w + o); o += 64 * 4;
  int* counts = (int*)(w + o); o += 8 * 4;
  int* cursor = (int*)(w + o); o += 4 * 4;
  int* segoff = (int*)(w + o); o += 8 * 4;
  int* lang_e = (int*)(w + o); o += 2 * 4;
  float* cross = qbuf;  // q is dead after attention

  const long long sW13 = (long long)2816 * 1024;
  const long long sW2  = (long long)1024 * 2816;

  init_kernel<<<18, 256, 0, stream>>>(perm, counts, cursor);
  lang_gate_kernel<<<1, 256, 0, stream>>>(language, lang_emb, lgate_w, lgate_b,
                                          lang_e, tileL, counts);
  // first experts (per-batch expert): h = silu(x W1^T) * (x W3^T)  [bf16 h]
  gemm_kernel<true, 1, false, false><<<dim3(44, 32), 256, 0, stream>>>(
      x, lw1, lw3, nullptr, nullptr, h_buf, 4096, 2816, 1024, 2816,
      nullptr, tileL, sW13);
  gemm_kernel<false, 0, false, true><<<dim3(16, 32), 256, 0, stream>>>(
      h_buf, lw2, nullptr, nullptr, nullptr, first, 4096, 1024, 2816, 1024,
      nullptr, tileL, sW2);
  // cross attention
  gemm_kernel<false, 0, false, false><<<dim3(16, 32), 256, 0, stream>>>(
      first, in_w, nullptr, in_b, nullptr, qbuf, 4096, 1024, 1024, 1024,
      nullptr, nullptr, 0);
  gemm_kernel<false, 0, false, false><<<dim3(32, 2), 256, 0, stream>>>(
      caption, in_w + (size_t)1024 * 1024, nullptr, in_b + 1024, nullptr, kvb,
      154, 2048, 1024, 2048, nullptr, nullptr, 0);
  attn_kernel<<<4096, 256, 0, stream>>>(qbuf, kvb, attob);
  gemm_kernel<false, 0, false, false><<<dim3(16, 32), 256, 0, stream>>>(
      attob, out_w, nullptr, out_b, nullptr, cross, 4096, 1024, 1024, 1024,
      nullptr, nullptr, 0);
  // per-token caption routing -> grouped GEMM over padded expert segments
  cap_gate_kernel<<<512, 256, 0, stream>>>(cross, cgate_w, cgate_b, cap_e, counts);
  prep_kernel<<<1, 1, 0, stream>>>(counts, segoff, tileC);
  scatter_kernel<<<16, 256, 0, stream>>>(cap_e, segoff, cursor, perm);
  gemm_kernel<true, 1, true, false><<<dim3(44, 36), 256, 0, stream>>>(
      first, cw1, cw3, nullptr, nullptr, h_buf, 4608, 2816, 1024, 2816,
      perm, tileC, sW13);
  gemm_kernel<false, 2, false, true><<<dim3(16, 36), 256, 0, stream>>>(
      h_buf, cw2, nullptr, nullptr, first, out, 4608, 1024, 2816, 1024,
      perm, tileC, sW2);
  loss_kernel<<<1, 32, 0, stream>>>(counts, out + (size_t)4096 * 1024);
}
